// Block_16054587752739
// MI455X (gfx1250) — matrix operand order repaired and checked by reading, not compiled, not run on hardware
//
#include <hip/hip_runtime.h>

typedef __attribute__((ext_vector_type(16))) _Float16 v16h;
typedef __attribute__((ext_vector_type(8)))  float    v8f;

#define TT 256
#define CE 384
#define NH 6
#define HD 64

// ---------------------------------------------------------------------------
// WMMA helpers (CDNA5 gfx1250, wave32): D = A(16x32 f16) x B(32x16 f16) + C(f32)
// ---------------------------------------------------------------------------
__device__ __forceinline__ v8f wmma_f16(v16h a, v16h b, v8f c) {
  return __builtin_amdgcn_wmma_f32_16x16x32_f16(
      /*neg_a=*/false, a, /*neg_b=*/false, b,
      /*c_mod=*/(short)0, c, /*reuse_a=*/false, /*reuse_b=*/false);
}

// A-fragment, 16x32 f16, row-major source with leading dim `ld` (halves).
// ISA layout: lane M = lane%16, half = lane/16;
//   VGPR v in 0..3 -> K = half*8 + 2v,2v+1 ; VGPR v in 4..7 -> K = 16 + half*8 + ...
__device__ __forceinline__ v16h load_a16(const _Float16* base, int ld, int row,
                                         int koff, int lane) {
  union { v16h h; unsigned int u[8]; } f;
  const unsigned int* p =
      (const unsigned int*)(base + (size_t)row * ld + koff);
  int h4 = (lane >> 4) * 4;
#pragma unroll
  for (int v = 0; v < 4; ++v) f.u[v] = p[h4 + v];
#pragma unroll
  for (int v = 0; v < 4; ++v) f.u[4 + v] = p[8 + h4 + v];
  return f.h;
}

// B-fragment, 32x16 f16, from transposed operand Bt[n][k] with leading dim `ld`.
// ISA layout: lane N = lane%16, half = lane/16; elements 0..7 -> K = half*8 + 0..7,
// elements 8..15 -> K = 16 + half*8 + 0..7 (same order as A) -> two b128 loads.
__device__ __forceinline__ v16h load_b16(const _Float16* baseT, int ld, int col,
                                         int koff, int lane) {
  union { v16h h; uint4 q[2]; } f;
  const uint4* p =
      (const uint4*)(baseT + (size_t)col * ld + koff + ((lane >> 4) << 3));
  f.q[0] = p[0];
  f.q[1] = p[2];
  return f.h;
}

// ---------------------------------------------------------------------------
// WMMA GEMM, 16x64 tile per wave (register-blocked: A fragment reused by 4 WMMAs)
// C(MxN) = A(MxK,f16 rowmajor) * Bt(NxK,f16)^T
// epilogue: *alpha, +bias, +residual, relu, multiple output layouts
// mode 0: f32 rowmajor   mode 1: f16 rowmajor
// mode 2: f16 [B][H][T][D] head layout      mode 3: f16 [B][H][D][T] transposed
// Requires N % 64 == 0 (true for 384 and 1536).
// ---------------------------------------------------------------------------
__global__ __launch_bounds__(64) void wmma_gemm_kernel(
    const _Float16* __restrict__ A, const _Float16* __restrict__ Bt,
    const float* __restrict__ bias, const float* __restrict__ resid,
    float* __restrict__ outF, _Float16* __restrict__ outH, int M, int N, int K,
    int mode, int relu, float alpha) {
  int lane = threadIdx.x & 31;
  int wave = threadIdx.x >> 5;
  int mbase = blockIdx.x * 16;
  int nbase = (blockIdx.y * 2 + wave) * 64;
  if (mbase >= M || nbase >= N) return;

  int arow = mbase + (lane & 15);
  int bcol = nbase + (lane & 15);
  v8f acc0 = {}, acc1 = {}, acc2 = {}, acc3 = {};
  for (int ks = 0; ks < K; ks += 32) {
    if (ks + 32 < K)  // prefetch next A chunk (global_prefetch_b8)
      __builtin_prefetch(A + (size_t)arow * K + ks + 32, 0, 1);
    v16h a  = load_a16(A, K, arow, ks, lane);
    v16h b0 = load_b16(Bt, K, bcol +  0, ks, lane);
    v16h b1 = load_b16(Bt, K, bcol + 16, ks, lane);
    v16h b2 = load_b16(Bt, K, bcol + 32, ks, lane);
    v16h b3 = load_b16(Bt, K, bcol + 48, ks, lane);
    acc0 = wmma_f16(a, b0, acc0);
    acc1 = wmma_f16(a, b1, acc1);
    acc2 = wmma_f16(a, b2, acc2);
    acc3 = wmma_f16(a, b3, acc3);
  }

  int half = lane >> 4;
  int l16 = lane & 15;
#pragma unroll
  for (int c = 0; c < 4; ++c) {
    v8f acc = (c == 0) ? acc0 : (c == 1) ? acc1 : (c == 2) ? acc2 : acc3;
    int n = nbase + c * 16 + l16;
    float bv = bias ? bias[n] : 0.0f;
#pragma unroll
    for (int r = 0; r < 8; ++r) {
      int m = mbase + r + 8 * half;
      float v = acc[r] * alpha + bv;
      if (resid) v += resid[(size_t)m * N + n];
      if (relu) v = v > 0.0f ? v : 0.0f;
      if (mode == 0) {
        outF[(size_t)m * N + n] = v;
      } else if (mode == 1) {
        outH[(size_t)m * N + n] = (_Float16)v;
      } else {
        int b_ = m >> 8;   // m / T (T==256)
        int t_ = m & 255;
        int h_ = n >> 6;   // n / D (D==64)
        int d_ = n & 63;
        size_t idx;
        if (mode == 2)
          idx = (((size_t)(b_ * NH + h_) * TT + t_) * HD + d_);
        else
          idx = (((size_t)(b_ * NH + h_) * HD + d_) * TT + t_);
        outH[idx] = (_Float16)v;
      }
    }
  }
}

// ---------------------------------------------------------------------------
// LayerNorm (C=384): one wave per row, emits f16 activations for WMMA A operand
// ---------------------------------------------------------------------------
__global__ __launch_bounds__(256) void ln_f16_kernel(
    const float* __restrict__ x, const float* __restrict__ g,
    const float* __restrict__ be, _Float16* __restrict__ out, int rows) {
  int row = blockIdx.x * 8 + (threadIdx.x >> 5);
  int lane = threadIdx.x & 31;
  if (row >= rows) return;
  const float* xr = x + (size_t)row * CE;
  float v[12];
  float s1 = 0.f, s2 = 0.f;
#pragma unroll
  for (int j = 0; j < 12; ++j) {
    v[j] = xr[lane + j * 32];
    s1 += v[j];
    s2 += v[j] * v[j];
  }
#pragma unroll
  for (int w = 1; w < 32; w <<= 1) {
    s1 += __shfl_xor(s1, w, 32);
    s2 += __shfl_xor(s2, w, 32);
  }
  float mu = s1 * (1.0f / CE);
  float var = s2 * (1.0f / CE) - mu * mu;
  float rs = rsqrtf(var + 1e-5f);
  _Float16* orow = out + (size_t)row * CE;
#pragma unroll
  for (int j = 0; j < 12; ++j) {
    int c = lane + j * 32;
    orow[c] = (_Float16)((v[j] - mu) * rs * g[c] + be[c]);
  }
}

// Weight transpose + f32->f16 cast: Wt[n][k] = W[k][n]
__global__ __launch_bounds__(256) void wtrans_kernel(
    const float* __restrict__ W, _Float16* __restrict__ Wt, int K, int N) {
  int idx = blockIdx.x * 256 + threadIdx.x;
  if (idx >= K * N) return;
  int n = idx / K, k = idx % K;
  Wt[idx] = (_Float16)W[(size_t)k * N + n];
}

// ---------------------------------------------------------------------------
// Causal flash attention: one wave per (b, h, 16 query rows).
// q,k: f16 [B][H][T][D];  vt: f16 [B][H][D][T];  att out: f16 [B*T][C] rowmajor
// Q pre-scaled by D^-0.5 in the Q-projection GEMM.
// ---------------------------------------------------------------------------
__global__ __launch_bounds__(32) void attn_kernel(
    const _Float16* __restrict__ q, const _Float16* __restrict__ k,
    const _Float16* __restrict__ vt, _Float16* __restrict__ att) {
  __shared__ _Float16 Pl[16 * 32];  // P tile staging (C-layout -> A-layout)
  int lane = threadIdx.x & 31;
  int half = lane >> 4;
  int l16 = lane & 15;
  int bh = blockIdx.y;
  int b = bh / NH, h = bh % NH;
  int qbase = blockIdx.x * 16;

  const _Float16* qh = q + (size_t)bh * TT * HD;
  const _Float16* kh = k + (size_t)bh * TT * HD;
  const _Float16* vh = vt + (size_t)bh * HD * TT;

  v16h aq0 = load_a16(qh, HD, qbase + l16, 0, lane);
  v16h aq1 = load_a16(qh, HD, qbase + l16, 32, lane);

  v8f o0 = {}, o1 = {}, o2 = {}, o3 = {};
  float mrow[8], lrow[8];
#pragma unroll
  for (int r = 0; r < 8; ++r) { mrow[r] = -1e30f; lrow[r] = 0.f; }

  int nkb = qbase / 32 + 1;  // causal: only key blocks covering <= qbase+15
  for (int kb32 = 0; kb32 < nkb; ++kb32) {
    int kb = kb32 * 32;
    // S = Q * K^T  (two 16x16 tiles covering 32 keys; K-dim 64 = 2 steps)
    v8f s0 = {}, s1 = {};
    s0 = wmma_f16(aq0, load_b16(kh, HD, kb + l16, 0, lane), s0);
    s0 = wmma_f16(aq1, load_b16(kh, HD, kb + l16, 32, lane), s0);
    s1 = wmma_f16(aq0, load_b16(kh, HD, kb + 16 + l16, 0, lane), s1);
    s1 = wmma_f16(aq1, load_b16(kh, HD, kb + 16 + l16, 32, lane), s1);

    int key0 = kb + l16;
    int key1 = key0 + 16;
#pragma unroll
    for (int r = 0; r < 8; ++r) {
      int rowq = qbase + r + 8 * half;
      float e0 = (key0 <= rowq) ? s0[r] : -1e30f;
      float e1 = (key1 <= rowq) ? s1[r] : -1e30f;
      float t = fmaxf(e0, e1);
#pragma unroll
      for (int w = 1; w < 16; w <<= 1) t = fmaxf(t, __shfl_xor(t, w, 16));
      float mn = fmaxf(mrow[r], t);
      float a = __expf(mrow[r] - mn);
      float p0 = __expf(e0 - mn);
      float p1 = __expf(e1 - mn);
      float ps = p0 + p1;
#pragma unroll
      for (int w = 1; w < 16; w <<= 1) ps += __shfl_xor(ps, w, 16);
      lrow[r] = lrow[r] * a + ps;
      mrow[r] = mn;
      o0[r] *= a; o1[r] *= a; o2[r] *= a; o3[r] *= a;
      Pl[(r + 8 * half) * 32 + l16]      = (_Float16)p0;
      Pl[(r + 8 * half) * 32 + 16 + l16] = (_Float16)p1;
    }
    __syncthreads();
    // O += P(16x32) * V(32x64): A from LDS, B = vt rows (contiguous keys)
    v16h ap = load_a16(Pl, 32, l16, 0, lane);
    o0 = wmma_f16(ap, load_b16(vh, TT, 0 * 16 + l16, kb, lane), o0);
    o1 = wmma_f16(ap, load_b16(vh, TT, 1 * 16 + l16, kb, lane), o1);
    o2 = wmma_f16(ap, load_b16(vh, TT, 2 * 16 + l16, kb, lane), o2);
    o3 = wmma_f16(ap, load_b16(vh, TT, 3 * 16 + l16, kb, lane), o3);
    __syncthreads();
  }

#pragma unroll
  for (int r = 0; r < 8; ++r) {
    float inv = 1.0f / lrow[r];
    int m = b * TT + qbase + r + 8 * half;
    size_t base = (size_t)m * CE + h * HD;
    att[base + 0 * 16 + l16] = (_Float16)(o0[r] * inv);
    att[base + 1 * 16 + l16] = (_Float16)(o1[r] * inv);
    att[base + 2 * 16 + l16] = (_Float16)(o2[r] * inv);
    att[base + 3 * 16 + l16] = (_Float16)(o3[r] * inv);
  }
}

// ---------------------------------------------------------------------------
extern "C" void kernel_launch(void* const* d_in, const int* in_sizes, int n_in,
                              void* d_out, int out_size, void* d_ws,
                              size_t ws_size, hipStream_t stream) {
  (void)in_sizes; (void)n_in; (void)out_size; (void)ws_size;
  const float* x   = (const float*)d_in[0];
  const float* Wk  = (const float*)d_in[1];
  const float* Wq  = (const float*)d_in[2];
  const float* Wv  = (const float*)d_in[3];
  const float* Wo  = (const float*)d_in[4];
  const float* bo  = (const float*)d_in[5];
  const float* W1  = (const float*)d_in[6];
  const float* b1  = (const float*)d_in[7];
  const float* W2  = (const float*)d_in[8];
  const float* b2  = (const float*)d_in[9];
  const float* g1  = (const float*)d_in[10];
  const float* be1 = (const float*)d_in[11];
  const float* g2  = (const float*)d_in[12];
  const float* be2 = (const float*)d_in[13];

  const long M = 128L * 256;  // 32768 token rows
  char* ws = (char*)d_ws;
  size_t off = 0;
  // Persistent across both phases:
  _Float16* W1t = (_Float16*)(ws + off); off += 384L * 1536 * 2;
  _Float16* W2t = (_Float16*)(ws + off); off += 1536L * 384 * 2;
  float*    x1  = (float*)(ws + off);    off += (size_t)M * CE * 4;
  _Float16* h2  = (_Float16*)(ws + off); off += (size_t)M * CE * 2;
  char* big = ws + off;  // region reused: phase1 {h1,Wq/k/v/o^T,q,k,v,att} -> phase2 {ff1}

  size_t p = 0;
  _Float16* h1  = (_Float16*)(big + p); p += (size_t)M * CE * 2;       // 25.2 MB
  _Float16* Wqt = (_Float16*)(big + p); p += 384L * 384 * 2;
  _Float16* Wkt = (_Float16*)(big + p); p += 384L * 384 * 2;
  _Float16* Wvt = (_Float16*)(big + p); p += 384L * 384 * 2;
  _Float16* Wot = (_Float16*)(big + p); p += 384L * 384 * 2;
  _Float16* qh  = (_Float16*)(big + p); p += 128L * NH * TT * HD * 2;  // 25.2 MB
  _Float16* kh  = (_Float16*)(big + p); p += 128L * NH * TT * HD * 2;
  _Float16* vh  = (_Float16*)(big + p); p += 128L * NH * TT * HD * 2;
  _Float16* att = (_Float16*)(big + p); p += (size_t)M * CE * 2;
  _Float16* ff1 = (_Float16*)big;  // phase-2 reuse, 32768 x 1536 f16

  // Weight transpose+cast
  wtrans_kernel<<<(384 * 384 + 255) / 256, 256, 0, stream>>>(Wq, Wqt, 384, 384);
  wtrans_kernel<<<(384 * 384 + 255) / 256, 256, 0, stream>>>(Wk, Wkt, 384, 384);
  wtrans_kernel<<<(384 * 384 + 255) / 256, 256, 0, stream>>>(Wv, Wvt, 384, 384);
  wtrans_kernel<<<(384 * 384 + 255) / 256, 256, 0, stream>>>(Wo, Wot, 384, 384);
  wtrans_kernel<<<(384 * 1536 + 255) / 256, 256, 0, stream>>>(W1, W1t, 384, 1536);
  wtrans_kernel<<<(1536 * 384 + 255) / 256, 256, 0, stream>>>(W2, W2t, 1536, 384);

  // h1 = LN1(x) in f16
  ln_f16_kernel<<<M / 8, 256, 0, stream>>>(x, g1, be1, h1, (int)M);

  dim3 g384(M / 16, 384 / 128);    // 2048 x 3, 2 waves/block, 64 cols/wave
  dim3 g1536(M / 16, 1536 / 128);  // 2048 x 12

  // Q (scaled by D^-0.5), K -> [B][H][T][D];  V -> [B][H][D][T]
  wmma_gemm_kernel<<<g384, 64, 0, stream>>>(h1, Wqt, nullptr, nullptr, nullptr,
                                            qh, M, 384, 384, 2, 0, 0.125f);
  wmma_gemm_kernel<<<g384, 64, 0, stream>>>(h1, Wkt, nullptr, nullptr, nullptr,
                                            kh, M, 384, 384, 2, 0, 1.0f);
  wmma_gemm_kernel<<<g384, 64, 0, stream>>>(h1, Wvt, nullptr, nullptr, nullptr,
                                            vh, M, 384, 384, 3, 0, 1.0f);

  // Causal flash attention -> att (rowmajor f16 [B*T][384])
  attn_kernel<<<dim3(TT / 16, 128 * NH), 32, 0, stream>>>(qh, kh, vh, att);

  // x1 = x + att @ Wo + bo   (f32)
  wmma_gemm_kernel<<<g384, 64, 0, stream>>>(att, Wot, bo, x, x1, nullptr, M,
                                            384, 384, 0, 0, 1.0f);

  // h2 = LN2(x1) in f16
  ln_f16_kernel<<<M / 8, 256, 0, stream>>>(x1, g2, be2, h2, (int)M);

  // ff1 = relu(h2 @ W1 + b1)  (f16)
  wmma_gemm_kernel<<<g1536, 64, 0, stream>>>(h2, W1t, b1, nullptr, nullptr,
                                             ff1, M, 1536, 384, 1, 1, 1.0f);

  // out = x1 + ff1 @ W2 + b2  (f32)
  wmma_gemm_kernel<<<g384, 64, 0, stream>>>(ff1, W2t, b2, x1, (float*)d_out,
                                            nullptr, M, 384, 1536, 0, 0, 1.0f);
}
